// TaskAlignedAssigner_65790309040698
// MI455X (gfx1250) — matrix operand order repaired and checked by reading, not compiled, not run on hardware
//
#include <hip/hip_runtime.h>
#include <hip/hip_bf16.h>

typedef __attribute__((ext_vector_type(16))) _Float16 v16h;
typedef __attribute__((ext_vector_type(8)))  float    v8f;

#define N_ANCH 33600
#define M_GT   512
#define NCLS   80
#define TOPKK  10
#define NEGV   (-100000000.0f)
#define EPSV   (1e-9f)

__device__ __forceinline__ bool better(float v1, int n1, float v2, int n2) {
    return (v1 > v2) || (v1 == v2 && n1 < n2);
}
__device__ __forceinline__ float pow6(float x) { float x2 = x * x; return x2 * x2 * x2; }
__device__ __forceinline__ float iou_fn(float px0, float py0, float px1, float py1,
                                        float gx0, float gy0, float gx1, float gy1) {
    float iw = fmaxf(fminf(px1, gx1) - fmaxf(px0, gx0), 0.0f);
    float ih = fmaxf(fminf(py1, gy1) - fmaxf(py0, gy0), 0.0f);
    float inter = iw * ih;
    float w1 = fmaxf(px1 - px0, 0.0f), h1 = fmaxf(py1 - py0, 0.0f);
    float w2 = fmaxf(gx1 - gx0, 0.0f), h2 = fmaxf(gy1 - gy0, 0.0f);
    float uni = w1 * h1 + w2 * h2 - inter + 1e-7f;
    return inter / uni;
}

// ---------------- kernel 0: zero init outputs + per-anchor keys ----------------
__global__ void k_init(float* __restrict__ out, unsigned long long* __restrict__ keys) {
    int i = blockIdx.x * blockDim.x + threadIdx.x;
    int stride = gridDim.x * blockDim.x;
    const int totF = N_ANCH * 86;   // bboxes(4) + scores(80) + fg(1) + gt_idx(1)
    for (int p = i; p < totF; p += stride) out[p] = 0.0f;
    for (int p = i; p < N_ANCH; p += stride) keys[p] = 0ull;
}

// ---------------- kernel 1: WMMA one-hot gather + per-column top-10 ----------------
// grid = 32 blocks (16 GT columns each), 256 threads (8 waves), wave32.
__global__ __launch_bounds__(256) void k_topk(
    const float* __restrict__ scores,   // N x 80
    const float* __restrict__ pbox,     // N x 4
    const float* __restrict__ apts,     // N x 2
    const int*   __restrict__ labels,   // M
    const float* __restrict__ gbox,     // M x 4
    unsigned long long* __restrict__ keys,  // N (atomic argmax keys)
    float* __restrict__ metricMax)          // M
{
    __shared__ float s_vals[256 * TOPKK];
    __shared__ int   s_idxs[256 * TOPKK];

    const int tid  = threadIdx.x;
    const int lane = tid & 31;
    const int wv   = tid >> 5;
    const int j    = lane & 15;          // column within tile (B/C layout: N = lane%16)
    const bool hi  = lane >= 16;
    const int m0   = blockIdx.x * 16;
    const int g    = m0 + j;

    const float gx0 = gbox[g * 4 + 0], gy0 = gbox[g * 4 + 1];
    const float gx1 = gbox[g * 4 + 2], gy1 = gbox[g * 4 + 3];
    const int glab  = labels[g];

    float vals[TOPKK]; int idxs[TOPKK];
#pragma unroll
    for (int q = 0; q < TOPKK; q++) { vals[q] = -3.0e38f; idxs[q] = 0x7FFFFFFF; }

    const int nTiles = N_ANCH / 16;      // 2100
    for (int t = wv; t < nTiles; t += 8) {
        const int a0 = t * 16;
        const int na = a0 + j;           // anchor this lane stages & loads for A (A row M = lane%16)

        // prefetch next tile's rows into cache (global_prefetch_b8)
        int tnext = t + 8;
        if (tnext < nTiles) {
            __builtin_prefetch(scores + (long)(tnext * 16 + j) * NCLS, 0, 1);
            __builtin_prefetch(pbox + (long)(tnext * 16 + j) * 4, 0, 1);
        }

        const float4 pb = *(const float4*)(pbox + (long)na * 4);
        const float2 ap = *(const float2*)(apts + (long)na * 2);

        v8f acc;
#pragma unroll
        for (int q = 0; q < 8; q++) acc[q] = 0.0f;

        const float* srow = scores + (long)na * NCLS;
#pragma unroll
        for (int ks = 0; ks < 3; ++ks) {
            const int cbase = ks * 32;
            // A 16x32 f16 layout: lanes 0-15 halves = K {0..7, 16..23}; lanes 16-31 = K {8..15, 24..31}
            const int coff = cbase + (hi ? 8 : 0);
            float c1[8], c2[8];
#pragma unroll
            for (int q = 0; q < 8; q++) c1[q] = srow[coff + q];
            if (ks < 2) {
#pragma unroll
                for (int q = 0; q < 8; q++) c2[q] = srow[coff + 16 + q];
            } else {
#pragma unroll
                for (int q = 0; q < 8; q++) c2[q] = 0.0f;  // classes >= 80: one-hot rows are zero anyway
            }
            v16h ah, al, bm;
#pragma unroll
            for (int q = 0; q < 8; q++) {
                _Float16 h1 = (_Float16)c1[q];
                ah[q] = h1;  al[q] = (_Float16)(c1[q] - (float)h1);     // hi/lo split: exact gather
                _Float16 h2 = (_Float16)c2[q];
                ah[8 + q] = h2;  al[8 + q] = (_Float16)(c2[q] - (float)h2);
            }
            // B 32x16 f16 layout: lane%16 = column; lanes 0-15 K = {0..7, 16..23}, lanes 16-31 K = {8..15, 24..31}
            const int kb = cbase + (hi ? 8 : 0);
#pragma unroll
            for (int q = 0; q < 16; q++)
                bm[q] = (glab == kb + q + (q >= 8 ? 8 : 0)) ? (_Float16)1.0f : (_Float16)0.0f;

            acc = __builtin_amdgcn_wmma_f32_16x16x32_f16(false, ah, false, bm, (short)0, acc, false, false);
            acc = __builtin_amdgcn_wmma_f32_16x16x32_f16(false, al, false, bm, (short)0, acc, false, false);
        }

        // acc[r] = cls_score for (anchor row r + (hi?8:0), gt column j)
#pragma unroll
        for (int r = 0; r < 8; r++) {
            const int arow = r + (hi ? 8 : 0);
            const int n = a0 + arow;
            float px0 = __shfl(pb.x, arow, 32);
            float py0 = __shfl(pb.y, arow, 32);
            float px1 = __shfl(pb.z, arow, 32);
            float py1 = __shfl(pb.w, arow, 32);
            float ax  = __shfl(ap.x, arow, 32);
            float ay  = __shfl(ap.y, arow, 32);
            bool ing = (ax - gx0 > 0.0f) && (ay - gy0 > 0.0f) && (gx1 - ax > 0.0f) && (gy1 - ay > 0.0f);
            float iou = iou_fn(px0, py0, px1, py1, gx0, gy0, gx1, gy1);
            float cls = acc[r];
            float met = ing ? sqrtf(fmaxf(cls, 0.0f)) * pow6(fmaxf(iou, 0.0f)) : NEGV;
            if (better(met, n, vals[TOPKK - 1], idxs[TOPKK - 1])) {
                float cv = met; int cn = n;
#pragma unroll
                for (int q = 0; q < TOPKK; q++) {
                    bool sw = better(cv, cn, vals[q], idxs[q]);
                    float tv = vals[q]; int tn = idxs[q];
                    vals[q] = sw ? cv : tv;  idxs[q] = sw ? cn : tn;
                    cv = sw ? tv : cv;       cn = sw ? tn : cn;
                }
            }
        }
    }

    // merge: 16 lane-lists per column via LDS
#pragma unroll
    for (int q = 0; q < TOPKK; q++) { s_vals[tid * TOPKK + q] = vals[q]; s_idxs[tid * TOPKK + q] = idxs[q]; }
    __syncthreads();

    if (tid < 16) {
        float fv[TOPKK]; int fi[TOPKK];
#pragma unroll
        for (int q = 0; q < TOPKK; q++) { fv[q] = -3.0e38f; fi[q] = 0x7FFFFFFF; }
        for (int u = 0; u < 16; ++u) {
            const int src = u * 16 + tid;          // threads with tid%16 == this column
            for (int q = 0; q < TOPKK; q++) {
                float v = s_vals[src * TOPKK + q]; int n = s_idxs[src * TOPKK + q];
                if (!better(v, n, fv[TOPKK - 1], fi[TOPKK - 1])) break;  // source list sorted desc
                float cv = v; int cn = n;
#pragma unroll
                for (int q2 = 0; q2 < TOPKK; q2++) {
                    bool sw = better(cv, cn, fv[q2], fi[q2]);
                    float tv = fv[q2]; int tn = fi[q2];
                    fv[q2] = sw ? cv : tv;  fi[q2] = sw ? cn : tn;
                    cv = sw ? tv : cv;      cn = sw ? tn : cn;
                }
            }
        }
        const int gg = m0 + tid;
        metricMax[gg] = fmaxf(fv[0], EPSV);        // clip(column max, EPS)
        const float ggx0 = gbox[gg * 4 + 0], ggy0 = gbox[gg * 4 + 1];
        const float ggx1 = gbox[gg * 4 + 2], ggy1 = gbox[gg * 4 + 3];
        for (int q = 0; q < TOPKK; q++) {
            if (fv[q] > -1.0f) {                   // in_gts positives (metric >= 0); NEG entries filtered
                int n = fi[q];
                const float4 pbq = *(const float4*)(pbox + (long)n * 4);
                float iou = iou_fn(pbq.x, pbq.y, pbq.z, pbq.w, ggx0, ggy0, ggx1, ggy1);
                unsigned int ib = __float_as_uint(iou);   // iou >= 0 -> monotonic bits
                unsigned long long key =
                    ((unsigned long long)ib << 32) | (unsigned long long)(0xFFFFFFFFu - (unsigned)gg);
                atomicMax(&keys[n], key);          // max iou, ties -> lowest gt index
            }
        }
    }
}

// ---------------- kernel 2: per-anchor finalize ----------------
__global__ void k_final(const float* __restrict__ scores, const int* __restrict__ labels,
                        const float* __restrict__ gbox,
                        const unsigned long long* __restrict__ keys,
                        const float* __restrict__ metricMax, float* __restrict__ out)
{
    int n = blockIdx.x * blockDim.x + threadIdx.x;
    if (n >= N_ANCH) return;
    unsigned long long key = keys[n];
    if (key == 0ull) return;                       // not fg: zero-init already correct
    float iou = __uint_as_float((unsigned)(key >> 32));
    int g = (int)(0xFFFFFFFFu - (unsigned)key);
    int tcls = labels[g];
    float cls = scores[(long)n * NCLS + tcls];
    float met = sqrtf(fmaxf(cls, 0.0f)) * pow6(fmaxf(iou, 0.0f));
    float sc = met / metricMax[g] * iou;
    sc = fminf(fmaxf(sc, 0.0f), 1.0f);

    float* tb = out;                    // N x 4
    float* ts = out + N_ANCH * 4;       // N x 80
    float* fg = out + N_ANCH * 84;      // N
    float* tg = out + N_ANCH * 85;      // N
    tb[n * 4 + 0] = gbox[g * 4 + 0];
    tb[n * 4 + 1] = gbox[g * 4 + 1];
    tb[n * 4 + 2] = gbox[g * 4 + 2];
    tb[n * 4 + 3] = gbox[g * 4 + 3];
    ts[(long)n * NCLS + tcls] = sc;
    fg[n] = 1.0f;
    tg[n] = (float)g;
}

extern "C" void kernel_launch(void* const* d_in, const int* in_sizes, int n_in,
                              void* d_out, int out_size, void* d_ws, size_t ws_size,
                              hipStream_t stream) {
    const float* scores = (const float*)d_in[0];   // pred_scores  N x 80
    const float* pbox   = (const float*)d_in[1];   // pred_bboxes  N x 4
    const float* apts   = (const float*)d_in[2];   // anchor_points N x 2
    const int*   labels = (const int*)d_in[3];     // gt_labels    M
    const float* gboxp  = (const float*)d_in[4];   // gt_bboxes    M x 4
    float* out = (float*)d_out;

    unsigned long long* keys = (unsigned long long*)d_ws;                  // N * 8 bytes
    float* mmax = (float*)((char*)d_ws + (size_t)N_ANCH * 8);              // M * 4 bytes

    k_init<<<1024, 256, 0, stream>>>(out, keys);
    k_topk<<<M_GT / 16, 256, 0, stream>>>(scores, pbox, apts, labels, gboxp, keys, mmax);
    k_final<<<(N_ANCH + 255) / 256, 256, 0, stream>>>(scores, labels, gboxp, keys, mmax, out);
}
